// VQEmbedding_773094113562
// MI455X (gfx1250) — matrix operand order repaired and checked by reading, not compiled, not run on hardware
//
#include <hip/hip_runtime.h>
#include <cstdint>

// ---------------------------------------------------------------------------
// VQ-VAE nearest-codebook quantization for MI455X (gfx1250).
//   z_e_x: [64, 256, 32, 32] f32,  codebook: [1024, 256] f32
//   out  : (z_q_x, z_q_x) each [64, 256, 32, 32] f32, concatenated flat.
//
// GEMM (65536 x 1024 x 256) runs on v_wmma_f32_16x16x32_bf16 with an
// fp32->bf16 hi/lo split (3 WMMAs per 32-wide K step) for near-fp32 argmin
// accuracy at bf16 matrix throughput. A tiles live in registers (z read once
// from HBM); codebook tiles are double-buffered in LDS and filled with the
// CDNA5 async global->LDS path (GLOBAL_LOAD_ASYNC_TO_LDS_B128 / ASYNCcnt),
// overlapping the next tile's fill with the current tile's WMMAs.
// ---------------------------------------------------------------------------

typedef __attribute__((ext_vector_type(16))) __bf16 v16bf;
typedef __attribute__((ext_vector_type(8)))  float  v8f;

#define CDIM   256
#define KCODES 1024
#define HW     1024      // H*W per image
#define NPIX   65536     // B*H*W
#define OUTHALF 16777216 // 64*256*32*32
#define NCHUNK (KCODES / 16)

// ---------------------------------------------------------------------------
// Kernel 1: split codebook into bf16 hi/lo planes and compute ||c||^2.
// ---------------------------------------------------------------------------
__global__ void vq_prep_kernel(const float* __restrict__ cb,
                               __bf16* __restrict__ cbHi,
                               __bf16* __restrict__ cbLo,
                               float*  __restrict__ cnorm) {
  __shared__ float red[256];
  const int code = blockIdx.x;
  const int c    = threadIdx.x;
  const float v  = cb[code * CDIM + c];
  const __bf16 h = (__bf16)v;
  const __bf16 l = (__bf16)(v - (float)h);
  cbHi[code * CDIM + c] = h;
  cbLo[code * CDIM + c] = l;
  red[c] = v * v;
  __syncthreads();
  for (int s = 128; s > 0; s >>= 1) {
    if (c < s) red[c] += red[c + s];
    __syncthreads();
  }
  if (c == 0) cnorm[code] = red[0];
}

// Async 16-byte global -> LDS copy (CDNA5 direct path, tracked by ASYNCcnt).
// VGLOBAL async encoding: VDST = LDS byte-address VGPR, VADDR = 64-bit addr.
__device__ __forceinline__ void async_copy16(uint32_t ldsAddr, uint64_t gAddr) {
  asm volatile("global_load_async_to_lds_b128 %0, %1, off"
               :: "v"(ldsAddr), "v"(gAddr) : "memory");
}
__device__ __forceinline__ void wait_async0() {
  asm volatile("s_wait_asynccnt 0" ::: "memory");
}

// ---------------------------------------------------------------------------
// Kernel 2: WMMA distance GEMM + running argmin.
// 1024 blocks x 128 threads (4 waves). Each wave owns a 16-row M tile
// (rows stay inside one image b, so channel loads are fully coalesced).
// ---------------------------------------------------------------------------
__launch_bounds__(128, 1)
__global__ void vq_argmin_kernel(const float* __restrict__ z,
                                 const __bf16* __restrict__ cbHi,
                                 const __bf16* __restrict__ cbLo,
                                 const float* __restrict__ cnorm,
                                 int* __restrict__ idxOut) {
  // Double-buffered B tile: 2 x (16 codes x 256 channels) hi+lo = 32 KB.
  __shared__ __align__(32) __bf16 BtHi[2][16 * CDIM];
  __shared__ __align__(32) __bf16 BtLo[2][16 * CDIM];
  __shared__ float redD[4][16][16];
  __shared__ int   redI[4][16][16];

  const int t    = threadIdx.x;
  const int wv   = t >> 5;        // wave 0..3
  const int lane = t & 31;
  const int half = lane >> 4;     // 0: lanes 0-15, 1: lanes 16-31
  const int l16  = lane & 15;

  const int row0     = blockIdx.x * 64 + wv * 16;  // wave's first pixel row
  const int b        = row0 >> 10;                 // image index (rows never straddle)
  const int rbase    = row0 & 1023;
  const size_t zbase = (size_t)b * (CDIM * HW);

  // Cooperative-fill assignment: 4096 halfs per plane, 128 threads -> 64 B each.
  const int rowB = t >> 3;            // 0..15 (code row within tile)
  const int koff = (t & 7) * 32;      // 0..224 halfs (64 B granules)

  // ---- Load the wave's 16x256 A tile into registers as bf16 hi/lo -------
  // A fragment (16x32), packed here in contiguous k order: lanes L and L+16 hold row M=L%16;
// element e of the v16bf maps to K-offset e + (half ? 16 : 0).
  v16bf aHi[8], aLo[8];
  const int r = rbase + l16;       // this lane's pixel within image b
  float xnPart = 0.f;
#pragma unroll
  for (int kk = 0; kk < 8; ++kk) {
    const int c0 = kk * 32;
#pragma unroll
    for (int e = 0; e < 16; ++e) {
      const int ko = e + (half ? 16 : 0);
      const float val = z[zbase + (size_t)(c0 + ko) * HW + r];
      const __bf16 h = (__bf16)val;
      aHi[kk][e] = h;
      aLo[kk][e] = (__bf16)(val - (float)h);
      xnPart += val * val;
    }
  }
  // Lanes L and L^16 hold the two K-halves of the same row -> full ||x||^2.
  const float xnRow = xnPart + __shfl_xor(xnPart, 16, 32);
  // C/D layout: lane half==0 holds M = r8, half==1 holds M = 8 + r8.
  float xn[8];
#pragma unroll
  for (int r8 = 0; r8 < 8; ++r8)
    xn[r8] = __shfl(xnRow, r8 + (half ? 8 : 0), 32);

  float bestD[8];
  int   bestI[8];
#pragma unroll
  for (int i = 0; i < 8; ++i) { bestD[i] = 3.4e38f; bestI[i] = 0; }

  const int ncol  = l16;              // lane's code column inside the chunk
  const int kbase = half ? 16 : 0;    // B fragment K-half for this lane

  // Issue this thread's slice of the tile fill for `chunk` into buffer `buf`.
  auto issue_fill = [&](int chunk, int buf) {
    const size_t g = (size_t)(chunk * 16 + rowB) * CDIM + koff;
    const uint32_t lH = (uint32_t)(uintptr_t)(&BtHi[buf][rowB * CDIM + koff]);
    const uint32_t lL = (uint32_t)(uintptr_t)(&BtLo[buf][rowB * CDIM + koff]);
    const uint64_t gH = (uint64_t)(uintptr_t)(cbHi + g);
    const uint64_t gL = (uint64_t)(uintptr_t)(cbLo + g);
#pragma unroll
    for (int q = 0; q < 4; ++q) {
      async_copy16(lH + 16u * q, gH + 16u * q);
      async_copy16(lL + 16u * q, gL + 16u * q);
    }
  };

  // Prologue: fill buffer 0 with chunk 0.
  issue_fill(0, 0);
  wait_async0();
  __syncthreads();

  int buf = 0;
  for (int chunk = 0; chunk < NCHUNK; ++chunk) {
    const int code0 = chunk * 16;

    // Overlap: async-fill the next chunk into the other buffer, and pull
    // the chunk after that toward L2 while the WMMAs run.
    if (chunk + 1 < NCHUNK) issue_fill(chunk + 1, buf ^ 1);
    if (chunk + 2 < NCHUNK)
      __builtin_prefetch(cbHi + (size_t)(code0 + 32 + rowB) * CDIM, 0, 1);

    v8f acc = {0.f, 0.f, 0.f, 0.f, 0.f, 0.f, 0.f, 0.f};
#pragma unroll
    for (int kk = 0; kk < 8; ++kk) {
      const int c0 = kk * 32;
      // B fragment (32x16 bf16): lanes 0-15 K=0..15, lanes 16-31 K=16..31,
      // 16 contiguous halfs per lane -> two ds_load_b128.
      const v16bf bHi = *(const v16bf*)(&BtHi[buf][ncol * CDIM + c0 + kbase]);
      const v16bf bLo = *(const v16bf*)(&BtLo[buf][ncol * CDIM + c0 + kbase]);
      acc = __builtin_amdgcn_wmma_f32_16x16x32_bf16(false, aHi[kk], false, bHi,
                                                    (short)0, acc, false, false);
      acc = __builtin_amdgcn_wmma_f32_16x16x32_bf16(false, aHi[kk], false, bLo,
                                                    (short)0, acc, false, false);
      acc = __builtin_amdgcn_wmma_f32_16x16x32_bf16(false, aLo[kk], false, bHi,
                                                    (short)0, acc, false, false);
    }

    const int code = code0 + ncol;
    const float cn = cnorm[code];
#pragma unroll
    for (int r8 = 0; r8 < 8; ++r8) {
      const float d = xn[r8] - 2.f * acc[r8] + cn;
      // first-min tie-break to match jnp.argmin
      if (d < bestD[r8] || (d == bestD[r8] && code < bestI[r8])) {
        bestD[r8] = d;
        bestI[r8] = code;
      }
    }

    // Next buffer must be fully resident, and all waves done with this one.
    wait_async0();
    __syncthreads();
    buf ^= 1;
  }

  // ---- Cross-lane argmin reduce: 16 candidate lanes per row -------------
#pragma unroll
  for (int r8 = 0; r8 < 8; ++r8) {
    const int rowm = r8 + (half ? 8 : 0);
    redD[wv][rowm][l16] = bestD[r8];
    redI[wv][rowm][l16] = bestI[r8];
  }
  __syncthreads();
  if (lane < 16) {
    float bd = redD[wv][lane][0];
    int   bi = redI[wv][lane][0];
#pragma unroll
    for (int c = 1; c < 16; ++c) {
      const float d = redD[wv][lane][c];
      const int   i2 = redI[wv][lane][c];
      if (d < bd || (d == bd && i2 < bi)) { bd = d; bi = i2; }
    }
    idxOut[row0 + lane] = bi;
  }
}

// ---------------------------------------------------------------------------
// Kernel 3: gather original f32 codebook rows, write both tuple outputs.
// One block per (b, c) pair -> coalesced 1024-float rows.
// ---------------------------------------------------------------------------
__global__ void vq_scatter_kernel(const float* __restrict__ cb,
                                  const int* __restrict__ idx,
                                  float* __restrict__ out) {
  const int bc = blockIdx.x;     // b*256 + c
  const int b  = bc >> 8;
  const int c  = bc & 255;
  const size_t obase = (size_t)bc * HW;
#pragma unroll
  for (int rr = 0; rr < 4; ++rr) {
    const int r  = threadIdx.x + rr * 256;
    const int id = idx[b * HW + r];
    const float v = cb[id * CDIM + c];
    out[obase + r]           = v;
    out[OUTHALF + obase + r] = v;
  }
}

// ---------------------------------------------------------------------------
extern "C" void kernel_launch(void* const* d_in, const int* in_sizes, int n_in,
                              void* d_out, int out_size, void* d_ws, size_t ws_size,
                              hipStream_t stream) {
  const float* z  = (const float*)d_in[0];   // [64,256,32,32]
  const float* cb = (const float*)d_in[1];   // [1024,256]
  float* out = (float*)d_out;

  // Workspace layout (~1.3 MB):
  //   [0, 512K)        codebook hi  (bf16, 1024*256)
  //   [512K, 1M)       codebook lo  (bf16)
  //   [1M, 1M+4K)      ||c||^2      (f32, 1024)
  //   [1M+4K, +256K)   argmin index (i32, 65536)
  char* ws = (char*)d_ws;
  __bf16* cbHi = (__bf16*)(ws);
  __bf16* cbLo = (__bf16*)(ws + 524288);
  float*  cn   = (float*)(ws + 1048576);
  int*    idx  = (int*)(ws + 1052672);

  vq_prep_kernel<<<KCODES, 256, 0, stream>>>(cb, cbHi, cbLo, cn);
  vq_argmin_kernel<<<NPIX / 64, 128, 0, stream>>>(z, cbHi, cbLo, cn, idx);
  vq_scatter_kernel<<<64 * 256, 256, 0, stream>>>(cb, idx, out);
}
